// AdaptiveLiquidLayer_71725953843247
// MI455X (gfx1250) — compile-verified, matrix operand order checked by reading, not run on hardware
//
#include <hip/hip_runtime.h>
#include <hip/hip_bf16.h>

// Fused LiquidCell RK4 for MI455X (gfx1250, wave32, WMMA).
//
// Roofline: 168 MB of mandatory HBM traffic vs ~5 GFLOP -> memory bound
// (~7.2 us at 23.3 TB/s). GEMM is pushed onto v_wmma_f32_16x16x32_bf16
// (bf16 A/B, f32 accumulate) so the matrix pipe is ~40x under-subscribed
// and the kernel stays on the bandwidth roof. Everything is one pass:
// x tile + W staged to LDS as bf16, WMMA -> f32 input_current in VGPRs,
// RK4 (4x sigmoid) fused on top, single store of out.

typedef __attribute__((ext_vector_type(16))) __bf16 v16bf;
typedef __attribute__((ext_vector_type(8)))  float  v8f;

#define NTOT   256    // neurons (N)
#define KTOT   128    // input size (I)
#define BTILE  128    // batch rows per block (8 waves x 16 rows)
#define KSTR   136    // padded LDS row stride in bf16 elems (16B-aligned rows,
                      // stride%64 != 0 -> no systematic bank conflicts)

struct bh4 { __bf16 a, b, c, d; };           // 8-byte packed bf16 quad
union FragU { float4 f4[2]; v16bf v; };      // 32B fragment reinterpret

__device__ __forceinline__ float liquid_ode(float s, float icb, float wm,
                                            float sg, float am, float itau) {
    // dx/dt = -s/tau + sigmoid(sigma*(Wx + w*s + b)) * (A - s)
    float total = fmaf(s, wm, icb);
    float e = __expf(-sg * total);                    // v_exp_f32 path
    float f = __builtin_amdgcn_rcpf(1.0f + e);        // v_rcp_f32
    return fmaf(f, am - s, -s * itau);
}

__global__ __launch_bounds__(256)
void liquid_rk4_wmma(const float* __restrict__ x,
                     const float* __restrict__ h,
                     const float* __restrict__ W_in,
                     const float* __restrict__ w_rec,
                     const unsigned char* __restrict__ mask,   // jax bool -> 1 byte
                     const float* __restrict__ bias,
                     const float* __restrict__ tau,
                     const float* __restrict__ amp,            // A
                     const float* __restrict__ sigma,
                     float* __restrict__ out)
{
    __shared__ __align__(16) __bf16 sW[NTOT * KSTR];   // W_in as bf16, [n][k]
    __shared__ __align__(16) __bf16 sX[BTILE * KSTR];  // x tile as bf16, [m][k]
    __shared__ float sWm[NTOT], sBias[NTOT], sSig[NTOT], sAmp[NTOT], sITau[NTOT];

    const int tid = threadIdx.x;
    const int bb  = blockIdx.x * BTILE;

    // ---- per-neuron params (blockDim == NTOT == 256) ----
    {
        int n = tid;
        sWm[n]   = w_rec[n] * (mask[n] ? 1.0f : 0.0f);
        sBias[n] = bias[n];
        sSig[n]  = sigma[n];
        sAmp[n]  = amp[n];
        sITau[n] = 1.0f / tau[n];
    }

    // ---- W_in (256x128 f32 = 8192 float4) -> LDS bf16, coalesced ----
    {
        const float4* Wv = reinterpret_cast<const float4*>(W_in);
        #pragma unroll
        for (int i = 0; i < 32; ++i) {
            int f   = tid + i * 256;          // 0..8191
            int row = f >> 5;                 // / (128/4)
            int kg  = (f & 31) << 2;
            float4 v = Wv[f];
            bh4 p = { (__bf16)v.x, (__bf16)v.y, (__bf16)v.z, (__bf16)v.w };
            *reinterpret_cast<bh4*>(&sW[row * KSTR + kg]) = p;
        }
    }
    // ---- x tile (128x128 f32 = 4096 float4) -> LDS bf16, coalesced ----
    {
        const float4* Xv = reinterpret_cast<const float4*>(x + (size_t)bb * KTOT);
        #pragma unroll
        for (int i = 0; i < 16; ++i) {
            int f   = tid + i * 256;          // 0..4095
            int row = f >> 5;
            int kg  = (f & 31) << 2;
            float4 v = Xv[f];
            bh4 p = { (__bf16)v.x, (__bf16)v.y, (__bf16)v.z, (__bf16)v.w };
            *reinterpret_cast<bh4*>(&sX[row * KSTR + kg]) = p;
        }
    }
    __syncthreads();

    const int lane = tid & 31;
    const int wave = tid >> 5;           // 0..7
    const int half = lane >> 4;          // 0/1 (lane group)
    const int r16  = lane & 15;
    const int gb0  = bb + wave * 16;     // this wave's global batch row base

    // ---- A fragments: 16x32 bf16 per K-step, ISA layout:
    //   lanes 0-15:  VGPR0-3 = K 0..7,  VGPR4-7 = K 16..23
    //   lanes 16-31: VGPR0-3 = K 8..15, VGPR4-7 = K 24..31
    v16bf afr[4];
    {
        const __bf16* abase = &sX[(wave * 16 + r16) * KSTR];
        #pragma unroll
        for (int kk = 0; kk < 4; ++kk) {
            int k0 = kk * 32 + half * 8;
            FragU u;
            u.f4[0] = *reinterpret_cast<const float4*>(abase + k0);
            u.f4[1] = *reinterpret_cast<const float4*>(abase + k0 + 16);
            afr[kk] = u.v;
        }
    }

    // Prefetch this wave's h rows (overlaps with the WMMA work below).
    #pragma unroll
    for (int v = 0; v < 8; ++v) {
        size_t idx = (size_t)(gb0 + v + half * 8) * NTOT + r16 * 16;
        __builtin_prefetch(&h[idx], 0, 1);    // global_prefetch_b8
    }

    // ---- per N-tile: GEMM (4x WMMA) then fused RK4 + store ----
    #pragma unroll
    for (int nt = 0; nt < 16; ++nt) {
        const int n = nt * 16 + r16;
        const __bf16* wbase = &sW[n * KSTR];

        v8f acc = {};
        #pragma unroll
        for (int kk = 0; kk < 4; ++kk) {
            int k0 = kk * 32 + half * 8;
            FragU u;
            u.f4[0] = *reinterpret_cast<const float4*>(wbase + k0);
            u.f4[1] = *reinterpret_cast<const float4*>(wbase + k0 + 16);
            acc = __builtin_amdgcn_wmma_f32_16x16x32_bf16(
                      false, afr[kk], false, u.v, (short)0, acc, false, false);
        }

        // Per-neuron params (broadcast within lane group; LDS resident)
        const float wm = sWm[n], bi = sBias[n], sg = sSig[n];
        const float am = sAmp[n], it = sITau[n];

        // C/D layout: VGPR v -> M = v + 8*half, N = r16 (+ nt*16)
        #pragma unroll
        for (int v = 0; v < 8; ++v) {
            const int m = v + half * 8;
            const size_t idx = (size_t)(gb0 + m) * NTOT + n;
            const float hv  = h[idx];
            const float icb = acc[v] + bi;

            float k1 = liquid_ode(hv,                 icb, wm, sg, am, it);
            float k2 = liquid_ode(fmaf(0.5f, k1, hv), icb, wm, sg, am, it);
            float k3 = liquid_ode(fmaf(0.5f, k2, hv), icb, wm, sg, am, it);
            float k4 = liquid_ode(hv + k3,            icb, wm, sg, am, it);

            out[idx] = fmaf(k1 + 2.0f * (k2 + k3) + k4, (1.0f / 6.0f), hv);
        }
    }
}

extern "C" void kernel_launch(void* const* d_in, const int* in_sizes, int n_in,
                              void* d_out, int out_size, void* d_ws, size_t ws_size,
                              hipStream_t stream) {
    const float*         x     = (const float*)d_in[0];
    const float*         h     = (const float*)d_in[1];
    const float*         W_in  = (const float*)d_in[2];
    const float*         w_rec = (const float*)d_in[3];
    const unsigned char* mask  = (const unsigned char*)d_in[4];
    const float*         bias  = (const float*)d_in[5];
    const float*         tau   = (const float*)d_in[6];
    const float*         amp   = (const float*)d_in[7];
    const float*         sigma = (const float*)d_in[8];
    float*               out   = (float*)d_out;

    const int B    = in_sizes[0] / KTOT;   // 65536
    const int grid = B / BTILE;            // 512 blocks of 256 threads

    liquid_rk4_wmma<<<grid, 256, 0, stream>>>(x, h, W_in, w_rec, mask,
                                              bias, tau, amp, sigma, out);
}
